// MultiHeadAttention_953482740172
// MI455X (gfx1250) — matrix operand order repaired and checked by reading, not compiled, not run on hardware
//
#include <hip/hip_runtime.h>
#include <hip/hip_bf16.h>
#include <stdint.h>
#include <stddef.h>

// ---------------------------------------------------------------------------
// Causal multi-head attention forward for MI455X (gfx1250, wave32, WMMA).
// Pipeline (all bf16 staging in d_ws, f32 accumulation in WMMA):
//   1) fp32 -> bf16 converts (x, Wq, Wk, Wv, Wo)
//   2) Q = x@Wq^T, K = x@Wk^T (bf16, row-major [B*S, D])
//   3) V = x@Wv^T stored transposed per head: Vt[b,h,dh,s]
//   4) flash attention (online softmax, causal) -> Z bf16 [B*S, D]
//   5) out = Z@Wo^T + bo (fp32)
// WMMA fragments are fetched as 2x b128 loads; the GEMM K-loop is manually
// ping-pong software-pipelined (two fragment buffers alternate roles, no
// rotation copies; loads for k+32 overlap WMMAs for k).
// ---------------------------------------------------------------------------

#define B_  4
#define S_  2048
#define D_  1024
#define H_  16
#define DH_ 64

typedef __attribute__((ext_vector_type(16))) __bf16       v16bf;
typedef __attribute__((ext_vector_type(8)))  float        v8f;
typedef __attribute__((ext_vector_type(4)))  unsigned int v4u;
typedef __attribute__((ext_vector_type(2)))  unsigned int v2u;

struct FragBF {
  union { v16bf bf; v4u q[2]; unsigned int u[8]; };
};

static __device__ __forceinline__ unsigned short f2bf(float f) {
  unsigned int u = __float_as_uint(f);
  u += 0x7FFFu + ((u >> 16) & 1u);           // round-to-nearest-even
  return (unsigned short)(u >> 16);
}
static __device__ __forceinline__ unsigned int pack2bf(float a, float b) {
  return (unsigned int)f2bf(a) | ((unsigned int)f2bf(b) << 16);
}

// A-fragment (16x32 bf16, M x K). ISA layout: lanes 0-15 hold M=lane with
// K in {0..7, 16..23}; lanes 16-31 hold same rows with K in {8..15, 24..31}.
// VGPRs 0-3 = 16 contiguous bytes at K=(hi*8), VGPRs 4-7 = 16 bytes at +32B.
static __device__ __forceinline__ FragBF load_a_frag(const unsigned short* base, size_t ld,
                                                     size_t m0, size_t k0, int lane) {
  FragBF f;
  const int hi = (lane >> 4) & 1;
  const unsigned short* p =
      base + (m0 + (size_t)(lane & 15)) * ld + k0 + (size_t)(hi * 8);
  f.q[0] = *(const v4u*)(p);
  f.q[1] = *(const v4u*)(p + 16);
  return f;
}

// B-fragment (32x16 bf16, K x N), element B[k][n] from memory laid out
// row-major [n, k]. Lanes 0-15: N=lane, K in {0..7, 16..23}; lanes 16-31:
// N=lane-16, K in {8..15, 24..31} -> per lane 2x b128 (same K order as A).
static __device__ __forceinline__ FragBF load_b_frag(const unsigned short* base, size_t ld,
                                                     size_t n0, size_t k0, int lane) {
  FragBF f;
  const int kb = ((lane >> 4) & 1) * 8;
  const unsigned short* p =
      base + (n0 + (size_t)(lane & 15)) * ld + k0 + (size_t)kb;
  f.q[0] = *(const v4u*)(p);
  f.q[1] = *(const v4u*)(p + 16);
  return f;
}

static __device__ __forceinline__ v8f wmma_bf16(const FragBF& a, const FragBF& b, v8f c) {
  // 8 args: (neg_a, A, neg_b, B, c_mod, C, reuse_a, reuse_b)
  return __builtin_amdgcn_wmma_f32_16x16x32_bf16(false, a.bf, false, b.bf,
                                                 (short)0, c, false, false);
}

// ---------------------------------------------------------------------------
// fp32 -> bf16 elementwise convert, 4 elements / thread, packed dword stores
// ---------------------------------------------------------------------------
__global__ __launch_bounds__(256) void k_f32_to_bf16(const float* __restrict__ src,
                                                     unsigned int* __restrict__ dst, int n4) {
  int i = blockIdx.x * blockDim.x + threadIdx.x;   // index in groups of 4 floats
  if (i < n4) {
    const float4 v = ((const float4*)src)[i];
    v2u o;
    o.x = pack2bf(v.x, v.y);
    o.y = pack2bf(v.z, v.w);
    *(v2u*)(dst + (size_t)i * 2) = o;
  }
}

// ---------------------------------------------------------------------------
// Y[M,N] = A[M,K] @ W[N,K]^T  (bf16 inputs, f32 WMMA accumulation)
// One wave computes a 16x64 tile (A-fragment reused across 4 B-fragments).
// Ping-pong pipelined K-loop: buffer 0 computes while buffer 1 loads k+32,
// then roles swap — no register rotation copies.
// mode 0: store bf16 row-major Y[m*N+n]
// mode 1: store bf16 transposed-per-head Vt[((b*H+h)*DH+dh)*S + s]
// mode 2: store fp32 Y[m*N+n] + bias[n]
// ---------------------------------------------------------------------------
__global__ __launch_bounds__(256) void k_gemm_nt_bf16(
    const unsigned short* __restrict__ A, const unsigned short* __restrict__ W,
    unsigned short* __restrict__ Ybf, float* __restrict__ Yf,
    const float* __restrict__ bias, int M, int N, int K, int mode) {
  const int lane   = threadIdx.x & 31;
  const int wid    = blockIdx.x * (blockDim.x >> 5) + (threadIdx.x >> 5);
  const int ntiles = N >> 6;
  const size_t m0  = (size_t)(wid / ntiles) * 16;
  const size_t n0  = (size_t)(wid % ntiles) * 64;
  if (m0 >= (size_t)M) return;

  const int hi = lane >> 4, nl = lane & 15;
  // per-lane invariant row bases (for prefetch)
  const unsigned short* pa = A + (m0 + (size_t)nl) * (size_t)K;
  const unsigned short* pw = W + (n0 + (size_t)nl) * (size_t)K;

  v8f acc[4] = {};

  FragBF a0, a1, b0[4], b1[4];
  a0 = load_a_frag(A, (size_t)K, m0, 0, lane);
#pragma unroll
  for (int t = 0; t < 4; ++t)
    b0[t] = load_b_frag(W, (size_t)K, n0 + (size_t)(16 * t), 0, lane);

  int k0 = 0;
  for (;;) {
    // --- phase 0: compute buf0 @ k0, load buf1 @ k0+32 -----------------
    int k1 = k0 + 32;
    if (k1 < K) {
      __builtin_prefetch(pa + k1 + 32, 0, 3);
      __builtin_prefetch(pw + k1 + 32, 0, 3);
      a1 = load_a_frag(A, (size_t)K, m0, (size_t)k1, lane);
#pragma unroll
      for (int t = 0; t < 4; ++t)
        b1[t] = load_b_frag(W, (size_t)K, n0 + (size_t)(16 * t), (size_t)k1, lane);
    }
#pragma unroll
    for (int t = 0; t < 4; ++t) acc[t] = wmma_bf16(a0, b0[t], acc[t]);
    if (k1 >= K) break;

    // --- phase 1: compute buf1 @ k1, load buf0 @ k1+32 -----------------
    const int k2 = k1 + 32;
    if (k2 < K) {
      __builtin_prefetch(pa + k2 + 32, 0, 3);
      __builtin_prefetch(pw + k2 + 32, 0, 3);
      a0 = load_a_frag(A, (size_t)K, m0, (size_t)k2, lane);
#pragma unroll
      for (int t = 0; t < 4; ++t)
        b0[t] = load_b_frag(W, (size_t)K, n0 + (size_t)(16 * t), (size_t)k2, lane);
    }
#pragma unroll
    for (int t = 0; t < 4; ++t) acc[t] = wmma_bf16(a1, b1[t], acc[t]);
    if (k2 >= K) break;
    k0 = k2;
  }

  if (mode == 0) {
#pragma unroll
    for (int t = 0; t < 4; ++t) {
      const size_t n = n0 + (size_t)(16 * t + nl);
#pragma unroll
      for (int r = 0; r < 8; ++r)
        Ybf[(m0 + (size_t)(r + hi * 8)) * (size_t)N + n] = f2bf(acc[t][r]);
    }
  } else if (mode == 2) {
#pragma unroll
    for (int t = 0; t < 4; ++t) {
      const size_t n  = n0 + (size_t)(16 * t + nl);
      const float  bv = bias[n];
#pragma unroll
      for (int r = 0; r < 8; ++r)
        Yf[(m0 + (size_t)(r + hi * 8)) * (size_t)N + n] = acc[t][r] + bv;
    }
  } else {  // mode 1: Vt[b,h,dh,s] — lane holds 8 consecutive tokens at fixed dh
    const size_t mbase = m0 + (size_t)(hi * 8);
    const size_t bb    = mbase / (size_t)S_;
    const size_t sloc  = mbase % (size_t)S_;
#pragma unroll
    for (int t = 0; t < 4; ++t) {
      const size_t n  = n0 + (size_t)(16 * t + nl);
      const size_t h  = n >> 6, dh = n & 63;
      const size_t o  = (((bb * H_ + h) * DH_ + dh) * (size_t)S_ + sloc);
      v4u pkt;
      pkt.x = pack2bf(acc[t][0], acc[t][1]);
      pkt.y = pack2bf(acc[t][2], acc[t][3]);
      pkt.z = pack2bf(acc[t][4], acc[t][5]);
      pkt.w = pack2bf(acc[t][6], acc[t][7]);
      *(v4u*)((unsigned int*)Ybf + (o >> 1)) = pkt;
    }
  }
}

// ---------------------------------------------------------------------------
// Flash attention (causal). One wave handles 16 query rows of one (b,h).
// S-tile of 32 keys per iteration: 4 WMMAs for Q*K^T (K=64 split 2x32),
// online softmax with shfl_xor row reductions (rows live in 16-lane halves),
// P restaged via per-wave LDS tile into A-fragment layout, 4 WMMAs for P*V.
// All fragment loads for a step are issued before the WMMAs consume them;
// the next KV tile is prefetched while the current one is processed.
// ---------------------------------------------------------------------------
__global__ __launch_bounds__(256) void k_flash_causal(
    const unsigned short* __restrict__ Q, const unsigned short* __restrict__ Kk,
    const unsigned short* __restrict__ Vt, unsigned short* __restrict__ Z) {
  __shared__ __align__(16) unsigned short pshm[8][16 * 32];

  const int lane = threadIdx.x & 31;
  const int w    = threadIdx.x >> 5;
  const int wid  = blockIdx.x * 8 + w;
  const int qt   = wid & (S_ / 16 - 1);       // 0..127
  const int h    = (wid >> 7) & (H_ - 1);
  const int b    = wid >> 11;
  const int q0   = qt * 16;
  const int hi   = lane >> 4, nl = lane & 15;

  const size_t qrow   = (size_t)b * S_ + (size_t)q0;
  const size_t krow0  = (size_t)b * S_;                       // K rows base
  const size_t vtrow0 = (size_t)((b * H_ + h) * DH_);         // Vt rows base
  const size_t hcol   = (size_t)h * 64;

  const FragBF aQ0 = load_a_frag(Q, (size_t)D_, qrow, hcol + 0, lane);
  const FragBF aQ1 = load_a_frag(Q, (size_t)D_, qrow, hcol + 32, lane);

  v8f   o[4] = {};
  float mi[8], li[8];
#pragma unroll
  for (int r = 0; r < 8; ++r) { mi[r] = -3.0e38f; li[r] = 0.0f; }

  for (int k0 = 0; k0 < q0 + 16; k0 += 32) {
    // prefetch next KV tile (one cacheline per lane spans the whole tile)
    if (k0 + 32 < q0 + 16) {
      __builtin_prefetch(Kk + (krow0 + (size_t)(k0 + 32 + lane)) * D_ + hcol, 0, 3);
      __builtin_prefetch(Vt + (vtrow0 + (size_t)lane) * S_ + (size_t)(k0 + 32), 0, 3);
      __builtin_prefetch(Vt + (vtrow0 + (size_t)(32 + lane)) * S_ + (size_t)(k0 + 32), 0, 3);
    }

    // ---- scores: S = Q K^T * 1/sqrt(DH), 16x32 tile -------------------
    // load all four K B-fragments first, then issue the 4 WMMAs
    FragBF bk[4];
    bk[0] = load_b_frag(Kk, (size_t)D_, krow0 + k0,      hcol,      lane);
    bk[1] = load_b_frag(Kk, (size_t)D_, krow0 + k0,      hcol + 32, lane);
    bk[2] = load_b_frag(Kk, (size_t)D_, krow0 + k0 + 16, hcol,      lane);
    bk[3] = load_b_frag(Kk, (size_t)D_, krow0 + k0 + 16, hcol + 32, lane);
    v8f s0 = {}, s1 = {};
    s0 = wmma_bf16(aQ0, bk[0], s0);
    s0 = wmma_bf16(aQ1, bk[1], s0);
    s1 = wmma_bf16(aQ0, bk[2], s1);
    s1 = wmma_bf16(aQ1, bk[3], s1);

    // ---- causal mask + scale ------------------------------------------
    float sv0[8], sv1[8];
#pragma unroll
    for (int r = 0; r < 8; ++r) {
      const int row = q0 + r + hi * 8;
      const int c0  = k0 + nl;
      const int c1  = k0 + 16 + nl;
      sv0[r] = (c0 <= row) ? s0[r] * 0.125f : -3.0e38f;
      sv1[r] = (c1 <= row) ? s1[r] * 0.125f : -3.0e38f;
    }

    // ---- online softmax update ----------------------------------------
#pragma unroll
    for (int r = 0; r < 8; ++r) {
      float vm = fmaxf(sv0[r], sv1[r]);
#pragma unroll
      for (int off = 8; off; off >>= 1) vm = fmaxf(vm, __shfl_xor(vm, off, 32));
      const float mnew  = fmaxf(mi[r], vm);
      const float alpha = __expf(mi[r] - mnew);
      const float p0 = __expf(sv0[r] - mnew);
      const float p1 = __expf(sv1[r] - mnew);
      float rs = p0 + p1;
#pragma unroll
      for (int off = 8; off; off >>= 1) rs += __shfl_xor(rs, off, 32);
      li[r] = li[r] * alpha + rs;
      mi[r] = mnew;
#pragma unroll
      for (int t = 0; t < 4; ++t) o[t][r] *= alpha;
      // stage P (row-major 16x32 bf16) into this wave's private LDS tile
      pshm[w][(r + hi * 8) * 32 + nl]      = f2bf(p0);
      pshm[w][(r + hi * 8) * 32 + 16 + nl] = f2bf(p1);
    }

    // ---- O += P V (A from LDS via ds_load_b128, B from transposed V) ---
    FragBF bv[4];
#pragma unroll
    for (int t = 0; t < 4; ++t)
      bv[t] = load_b_frag(Vt, (size_t)S_, vtrow0 + (size_t)(16 * t), (size_t)k0, lane);
    const FragBF aP = load_a_frag(&pshm[w][0], 32, 0, 0, lane);
#pragma unroll
    for (int t = 0; t < 4; ++t) o[t] = wmma_bf16(aP, bv[t], o[t]);
  }

  // ---- normalize + store Z bf16 row-major [B*S, D] ---------------------
  float inv[8];
#pragma unroll
  for (int r = 0; r < 8; ++r) inv[r] = 1.0f / li[r];
#pragma unroll
  for (int t = 0; t < 4; ++t) {
    const int dh = 16 * t + nl;
#pragma unroll
    for (int r = 0; r < 8; ++r) {
      const int row = r + hi * 8;
      Z[(qrow + (size_t)row) * D_ + hcol + (size_t)dh] = f2bf(o[t][r] * inv[r]);
    }
  }
}

// ---------------------------------------------------------------------------
extern "C" void kernel_launch(void* const* d_in, const int* in_sizes, int n_in,
                              void* d_out, int out_size, void* d_ws, size_t ws_size,
                              hipStream_t stream) {
  (void)in_sizes; (void)n_in; (void)out_size; (void)ws_size;
  const float* x  = (const float*)d_in[0];
  const float* Wq = (const float*)d_in[1];
  const float* Wk = (const float*)d_in[2];
  const float* Wv = (const float*)d_in[3];
  const float* Wo = (const float*)d_in[4];
  const float* bo = (const float*)d_in[5];

  const size_t NT = (size_t)B_ * S_ * D_;   // 8,388,608
  const size_t NW = (size_t)D_ * D_;        // 1,048,576

  // workspace layout (bf16/ushort elements), total ~88 MiB
  unsigned short* xb  = (unsigned short*)d_ws;
  unsigned short* wqb = xb  + NT;
  unsigned short* wkb = wqb + NW;
  unsigned short* wvb = wkb + NW;
  unsigned short* wob = wvb + NW;
  unsigned short* Qb  = wob + NW;
  unsigned short* Kb  = Qb  + NT;
  unsigned short* Vtb = Kb  + NT;
  unsigned short* Zb  = Vtb + NT;

  // 1) converts (4 floats per thread)
  k_f32_to_bf16<<<(int)(NT / 4 / 256), 256, 0, stream>>>(x,  (unsigned int*)xb,  (int)(NT / 4));
  k_f32_to_bf16<<<(int)(NW / 4 / 256), 256, 0, stream>>>(Wq, (unsigned int*)wqb, (int)(NW / 4));
  k_f32_to_bf16<<<(int)(NW / 4 / 256), 256, 0, stream>>>(Wk, (unsigned int*)wkb, (int)(NW / 4));
  k_f32_to_bf16<<<(int)(NW / 4 / 256), 256, 0, stream>>>(Wv, (unsigned int*)wvb, (int)(NW / 4));
  k_f32_to_bf16<<<(int)(NW / 4 / 256), 256, 0, stream>>>(Wo, (unsigned int*)wob, (int)(NW / 4));

  // 2-3) projections: waves = (M/16)*(N/64) = 8192; 8 waves / block
  const int M = B_ * S_;
  const int gemm_blocks = (M / 16) * (D_ / 64) / 8;   // 1024
  k_gemm_nt_bf16<<<gemm_blocks, 256, 0, stream>>>(xb, wqb, Qb,  nullptr, nullptr, M, D_, D_, 0);
  k_gemm_nt_bf16<<<gemm_blocks, 256, 0, stream>>>(xb, wkb, Kb,  nullptr, nullptr, M, D_, D_, 0);
  k_gemm_nt_bf16<<<gemm_blocks, 256, 0, stream>>>(xb, wvb, Vtb, nullptr, nullptr, M, D_, D_, 1);

  // 4) flash attention: B*H*(S/16) = 8192 wave-tiles, 8 waves / block
  const int fa_blocks = (B_ * H_ * (S_ / 16)) / 8;    // 1024
  k_flash_causal<<<fa_blocks, 256, 0, stream>>>(Qb, Kb, Vtb, Zb);

  // 5) output projection with bias -> fp32
  k_gemm_nt_bf16<<<gemm_blocks, 256, 0, stream>>>(Zb, wob, nullptr, (float*)d_out, bo, M, D_, D_, 2);
}
